// StackedRNNModel_9440338116931
// MI455X (gfx1250) — matrix operand order repaired and checked by reading, not compiled, not run on hardware
//
#include <hip/hip_runtime.h>

// Problem constants (match reference)
#define B_ 128
#define T_ 1024
#define D_ 64
#define H_ 256
#define O_ 128

typedef __attribute__((ext_vector_type(16))) __bf16 v16bf;
typedef __attribute__((ext_vector_type(8)))  float  v8f;

// ---------- helpers ----------
__device__ __forceinline__ unsigned short f2bf(float f) {
  union { float f; unsigned u; } c; c.f = f;
  unsigned u = c.u + 0x7FFFu + ((c.u >> 16) & 1u);   // round-to-nearest-even
  return (unsigned short)(u >> 16);
}

// B-fragment (or any pre-swizzled fragment): 32 lanes x 32B, lane-contiguous.
__device__ __forceinline__ v16bf ld_frag_pack(const unsigned short* frag, int lane) {
  union { uint4 u[2]; v16bf v; } r;
  const uint4* p = (const uint4*)(frag + lane * 16);
  r.u[0] = p[0]; r.u[1] = p[1];
  return r.v;
}

// A-fragment (16x32 bf16, M x K) from row-major bf16 [16][stride] in LDS.
// Contiguous-k layout (same as the B blobs): lanes 0-15 hold M=lane, K={kb..kb+15};
//             lanes 16-31 hold M=lane-16, K={kb+16..kb+31}.
__device__ __forceinline__ v16bf ld_frag_A(const unsigned short* base, int stride,
                                           int kbase, int lane) {
  int m = lane & 15, half = lane >> 4;
  const unsigned short* r = base + m * stride + kbase + half * 16;
  union { uint4 u[2]; v16bf v; } q;
  q.u[0] = *(const uint4*)(r);
  q.u[1] = *(const uint4*)(r + 8);
  return q.v;
}

// CDNA5 async global -> LDS copy (ASYNCcnt tracked)
__device__ __forceinline__ void async_b128(unsigned lds_off, const void* g) {
  unsigned long long ga = (unsigned long long)g;
  asm volatile("global_load_async_to_lds_b128 %0, %1, off"
               :: "v"(lds_off), "v"(ga) : "memory");
}
__device__ __forceinline__ void wait_async0() {
  asm volatile("s_wait_asynccnt 0" ::: "memory");
}

__device__ __forceinline__ float fast_tanh(float x) {
  // tanh(x) = (1 - e^-2x) / (1 + e^-2x);  e^-2x = 2^(-2*log2(e)*x)
  float e = __builtin_amdgcn_exp2f(-2.8853900817779268f * x);
  return (1.0f - e) * __builtin_amdgcn_rcpf(1.0f + e);
}

#define WMMA_BF16(A, Bv, C) \
  __builtin_amdgcn_wmma_f32_16x16x32_bf16(false, (A), false, (Bv), (short)0, (C), false, false)

// Pipelined 8-k-tile GEMM segment: A from row-major LDS, B from fragment blob
// (LDS or global). Prefetch depth 2; alternate accumulators to halve the
// WMMA->WMMA dependent chain.
__device__ __forceinline__ void gemm8(const unsigned short* abase, int astride,
                                      const unsigned short* bbase, int lane,
                                      v8f& acc0, v8f& acc1) {
  v16bf a0 = ld_frag_A(abase, astride, 0, lane);
  v16bf b0 = ld_frag_pack(bbase, lane);
  v16bf a1 = ld_frag_A(abase, astride, 32, lane);
  v16bf b1 = ld_frag_pack(bbase + 512, lane);
#pragma unroll
  for (int kt = 0; kt < 8; ++kt) {
    v16bf an = a1, bn = b1;
    if (kt + 2 < 8) {
      an = ld_frag_A(abase, astride, (kt + 2) * 32, lane);
      bn = ld_frag_pack(bbase + (size_t)(kt + 2) * 512, lane);
    }
    if (kt & 1) acc1 = WMMA_BF16(a0, b0, acc1);
    else        acc0 = WMMA_BF16(a0, b0, acc0);
    a0 = a1; b0 = b1; a1 = an; b1 = bn;
  }
}

// Same, but B fragments already live in registers (loop-invariant weights).
__device__ __forceinline__ void gemm8_regB(const unsigned short* abase, int astride,
                                           const v16bf* wb, int lane,
                                           v8f& acc0, v8f& acc1) {
  v16bf a0 = ld_frag_A(abase, astride, 0, lane);
  v16bf a1 = ld_frag_A(abase, astride, 32, lane);
#pragma unroll
  for (int kt = 0; kt < 8; ++kt) {
    v16bf an = a1;
    if (kt + 2 < 8) an = ld_frag_A(abase, astride, (kt + 2) * 32, lane);
    if (kt & 1) acc1 = WMMA_BF16(a0, wb[kt], acc1);
    else        acc0 = WMMA_BF16(a0, wb[kt], acc0);
    a0 = a1; a1 = an;
  }
}

// C-fragment layout: lanes 0-15: N=lane, M=vgpr(0..7); lanes 16-31: N=lane-16, M=8+vgpr
__device__ __forceinline__ void store_h_tanh(unsigned short* h, int stride, int nbase,
                                             int lane, v8f acc0, v8f acc1) {
  int n  = nbase + (lane & 15);
  int mb = (lane >> 4) * 8;
#pragma unroll
  for (int r = 0; r < 8; ++r)
    h[(mb + r) * stride + n] = f2bf(fast_tanh(acc0[r] + acc1[r]));
}

// ---------- prep kernel 1: x fp32 [B][T][D] -> bf16 time-major [T][B][D] ----------
__global__ void cvt_x_kernel(const float* __restrict__ x, unsigned short* __restrict__ xbf) {
  int i = blockIdx.x * 256 + threadIdx.x;
  if (i >= B_ * T_ * D_) return;
  int d = i & (D_ - 1);
  int t = (i >> 6) & (T_ - 1);
  int b = i >> 16;
  xbf[((size_t)t * B_ + b) * D_ + d] = f2bf(x[i]);
}

// ---------- prep kernel 2: weight fp32 [K][N] -> bf16 WMMA B-fragment blobs ----------
// Fragment f = nt*Kt + kt, 512 ushorts (1 KB) each. Within fragment:
// lane L stores 16 bf16: n = nt*16 + (L&15), K = kt*32 + (L>>4)*16 + {0..15} packed in pairs.
__global__ void swz_w_kernel(const float* __restrict__ W, unsigned short* __restrict__ dst,
                             int K, int N) {
  int i = blockIdx.x * 256 + threadIdx.x;
  int Kt = K >> 5;
  int nfrag = Kt * (N >> 4);
  if (i >= nfrag * 32) return;
  int lane = i & 31;
  int f = i >> 5;
  int kt = f % Kt, nt = f / Kt;
  int n  = nt * 16 + (lane & 15);
  int kb = kt * 32 + (lane >> 4) * 16;
  unsigned short tmp[16];
#pragma unroll
  for (int v = 0; v < 8; ++v) {
    tmp[2 * v]     = f2bf(W[(size_t)(kb + 2 * v)     * N + n]);
    tmp[2 * v + 1] = f2bf(W[(size_t)(kb + 2 * v + 1) * N + n]);
  }
  uint4* p = (uint4*)(dst + (size_t)f * 512 + lane * 16);
  p[0] = ((const uint4*)tmp)[0];
  p[1] = ((const uint4*)tmp)[1];
}

// ---------- LDS map (bytes, per workgroup) ----------
#define LDS_WH1   0          // 131072  (Wh1 B-fragments, all 16 n-tiles)
#define LDS_WX2   131072     // 131072  (Wx2 B-fragments)
#define LDS_H1    262144     // 2 x 8704 double-buffered h1 [16][272] bf16
#define LDS_H2    279552     // 2 x 8704 double-buffered h2
#define LDS_XB    296960     // 2 x 2304 double-buffered x_t [16][72] bf16
#define LDS_LGT   301568     // 16 x 136 fp32 logits
#define SMEM_SZ   310272
#define HSTRIDE   272        // bf16 elems; 544B row -> bank rotation by 8
#define XSTRIDE   72

// ---------- persistent RNN kernel: 8 WGs, one M-tile (16 batch rows) each ----------
__global__ __launch_bounds__(512) void rnn_persist(
    const unsigned short* __restrict__ xbf,
    const unsigned short* __restrict__ wx1f,
    const unsigned short* __restrict__ wh1f,
    const unsigned short* __restrict__ wx2f,
    const unsigned short* __restrict__ wh2f,
    const unsigned short* __restrict__ wdf,
    const float* __restrict__ b1, const float* __restrict__ b2,
    const float* __restrict__ bd, float* __restrict__ out) {
  extern __shared__ char smem[];
  const int tid = threadIdx.x, lane = tid & 31, wv = tid >> 5;
  const int mtile = blockIdx.x;  // 0..7, batch rows [16*mtile, 16*mtile+16)

  unsigned short* WH1L = (unsigned short*)(smem + LDS_WH1);
  unsigned short* WX2L = (unsigned short*)(smem + LDS_WX2);
#define H1P(bf) ((unsigned short*)(smem + LDS_H1 + (bf) * 8704))
#define H2P(bf) ((unsigned short*)(smem + LDS_H2 + (bf) * 8704))
#define XBP(bf) ((unsigned short*)(smem + LDS_XB + (bf) * 2304))
  float* LGT = (float*)(smem + LDS_LGT);

  // --- preload Wh1+Wx2 fragment blobs (256 KB, adjacent in ws) into LDS, async ---
  for (int c = tid; c < 16384; c += 512) {
    const char* g = (c < 8192) ? (const char*)wh1f + (size_t)c * 16
                               : (const char*)wx2f + (size_t)(c - 8192) * 16;
    async_b128((unsigned)(c * 16), g);
  }
  // --- zero both h1/h2 state buffers (34816 B) ---
  for (int c = tid; c < 8704; c += 512) ((unsigned*)(smem + LDS_H1))[c] = 0u;
  // --- stage x_0 slab (2 KB) into XB(0), async ---
  if (tid < 128) {
    const char* src = (const char*)xbf + (size_t)(mtile * 16) * (D_ * 2);
    async_b128((unsigned)(LDS_XB + (tid >> 3) * (XSTRIDE * 2) + (tid & 7) * 16),
               src + tid * 16);
  }

  // --- per-wave invariants: wave wv owns n-tile ni = wv (16 cols of H=256) ---
  const int ni = wv;
  const float b1v = b1[ni * 16 + (lane & 15)];
  const float b2v = b2[ni * 16 + (lane & 15)];
  // Register-resident weights for the whole 1024-step scan:
  //   Wx1 column (2 frags, 16 VGPRs) and Wh2 column (8 frags, 64 VGPRs).
  const v16bf wx1a = ld_frag_pack(wx1f + (size_t)(ni * 2 + 0) * 512, lane);
  const v16bf wx1b = ld_frag_pack(wx1f + (size_t)(ni * 2 + 1) * 512, lane);
  v16bf wh2r[8];
#pragma unroll
  for (int kt = 0; kt < 8; ++kt)
    wh2r[kt] = ld_frag_pack(wh2f + (size_t)(ni * 8 + kt) * 512, lane);

  wait_async0();
  __syncthreads();

  for (int t = 0; t < T_; ++t) {
    const int buf = t & 1, nb = buf ^ 1;

    // prefetch next x slab into the other buffer (overlaps layer compute)
    if (t + 1 < T_ && tid < 128) {
      const char* src = (const char*)xbf + ((size_t)(t + 1) * B_ + mtile * 16) * (D_ * 2);
      async_b128((unsigned)(LDS_XB + nb * 2304 + (tid >> 3) * (XSTRIDE * 2) + (tid & 7) * 16),
                 src + tid * 16);
    }

    // ---- layer 1: h1' = tanh(x_t @ Wx1 + h1 @ Wh1 + b1), this wave's 16 cols ----
    v8f acc0, acc1;
#pragma unroll
    for (int i = 0; i < 8; ++i) { acc0[i] = b1v; acc1[i] = 0.0f; }
    acc0 = WMMA_BF16(ld_frag_A(XBP(buf), XSTRIDE, 0, lane), wx1a, acc0);
    acc1 = WMMA_BF16(ld_frag_A(XBP(buf), XSTRIDE, 32, lane), wx1b, acc1);
    gemm8(H1P(buf), HSTRIDE, WH1L + (size_t)ni * 8 * 512, lane, acc0, acc1);
    store_h_tanh(H1P(nb), HSTRIDE, ni * 16, lane, acc0, acc1);
    __syncthreads();  // all of h1' visible before layer-2 consumes full K

    // ---- layer 2: h2' = tanh(h1' @ Wx2 + h2 @ Wh2 + b2) ----
    v8f c0, c1;
#pragma unroll
    for (int i = 0; i < 8; ++i) { c0[i] = b2v; c1[i] = 0.0f; }
    gemm8(H1P(nb), HSTRIDE, WX2L + (size_t)ni * 8 * 512, lane, c0, c1);
    gemm8_regB(H2P(buf), HSTRIDE, wh2r, lane, c0, c1);
    store_h_tanh(H2P(nb), HSTRIDE, ni * 16, lane, c0, c1);

    wait_async0();     // x_{t+1} staged
    __syncthreads();   // h2' visible; old buffers free to overwrite
  }

  // ---- final dense: logits = h2_T @ Wd + bd (final state in buffer 0; T even) ----
  if (wv < 8) {  // O=128 -> 8 n-tiles, waves 0..7
    const float bdv = bd[wv * 16 + (lane & 15)];
    v8f acc0, acc1;
#pragma unroll
    for (int i = 0; i < 8; ++i) { acc0[i] = bdv; acc1[i] = 0.0f; }
    gemm8(H2P(0), HSTRIDE, wdf + (size_t)wv * 8 * 512, lane, acc0, acc1);
    int n = wv * 16 + (lane & 15);
    int mb = (lane >> 4) * 8;
#pragma unroll
    for (int r = 0; r < 8; ++r) LGT[(mb + r) * 136 + n] = acc0[r] + acc1[r];
  }
  __syncthreads();

  // ---- row softmax (16 rows per WG; trivial one-time cost) ----
  if (tid < 16) {
    int r = tid;
    float mx = -3.0e38f;
    for (int c = 0; c < O_; ++c) mx = fmaxf(mx, LGT[r * 136 + c]);
    float s = 0.0f;
    for (int c = 0; c < O_; ++c) {
      float e = __builtin_amdgcn_exp2f((LGT[r * 136 + c] - mx) * 1.4426950408889634f);
      LGT[r * 136 + c] = e;
      s += e;
    }
    float inv = __builtin_amdgcn_rcpf(s);
    for (int c = 0; c < O_; ++c)
      out[(size_t)(mtile * 16 + r) * O_ + c] = LGT[r * 136 + c] * inv;
  }
}

// ---------- host launcher ----------
extern "C" void kernel_launch(void* const* d_in, const int* in_sizes, int n_in,
                              void* d_out, int out_size, void* d_ws, size_t ws_size,
                              hipStream_t stream) {
  (void)in_sizes; (void)n_in; (void)out_size; (void)ws_size;
  const float* x   = (const float*)d_in[0];
  const float* Wx1 = (const float*)d_in[1];
  const float* Wh1 = (const float*)d_in[2];
  const float* b1  = (const float*)d_in[3];
  const float* Wx2 = (const float*)d_in[4];
  const float* Wh2 = (const float*)d_in[5];
  const float* b2  = (const float*)d_in[6];
  const float* Wd  = (const float*)d_in[7];
  const float* bd  = (const float*)d_in[8];

  char* ws = (char*)d_ws;
  const size_t XBF  = 0;
  const size_t WX1F = XBF  + (size_t)T_ * B_ * D_ * 2;   // 16 MB of bf16 x
  const size_t WH1F = WX1F + (size_t)2 * 16 * 1024;       // Wh1 and Wx2 must be adjacent
  const size_t WX2F = WH1F + (size_t)8 * 16 * 1024;
  const size_t WH2F = WX2F + (size_t)8 * 16 * 1024;
  const size_t WDF  = WH2F + (size_t)8 * 16 * 1024;

  unsigned short* xbf  = (unsigned short*)(ws + XBF);
  unsigned short* wx1f = (unsigned short*)(ws + WX1F);
  unsigned short* wh1f = (unsigned short*)(ws + WH1F);
  unsigned short* wx2f = (unsigned short*)(ws + WX2F);
  unsigned short* wh2f = (unsigned short*)(ws + WH2F);
  unsigned short* wdf  = (unsigned short*)(ws + WDF);

  // prep: x transpose+convert, weight swizzles
  cvt_x_kernel<<<(B_ * T_ * D_ + 255) / 256, 256, 0, stream>>>(x, xbf);
  swz_w_kernel<<<(32  * 32 + 255) / 256, 256, 0, stream>>>(Wx1, wx1f,  64, 256);
  swz_w_kernel<<<(128 * 32 + 255) / 256, 256, 0, stream>>>(Wh1, wh1f, 256, 256);
  swz_w_kernel<<<(128 * 32 + 255) / 256, 256, 0, stream>>>(Wx2, wx2f, 256, 256);
  swz_w_kernel<<<(128 * 32 + 255) / 256, 256, 0, stream>>>(Wh2, wh2f, 256, 256);
  swz_w_kernel<<<(64  * 32 + 255) / 256, 256, 0, stream>>>(Wd,  wdf,  256, 128);

  hipFuncSetAttribute((const void*)rnn_persist,
                      hipFuncAttributeMaxDynamicSharedMemorySize, SMEM_SZ);
  rnn_persist<<<8, 512, SMEM_SZ, stream>>>(xbf, wx1f, wh1f, wx2f, wh2f, wdf,
                                           b1, b2, bd, (float*)d_out);
}
